// CrossFrameAttention_31078383354380
// MI455X (gfx1250) — matrix operand order repaired and checked by reading, not compiled, not run on hardware
//
#include <hip/hip_runtime.h>

typedef __attribute__((ext_vector_type(16))) _Float16     v16h;
typedef __attribute__((ext_vector_type(8)))  _Float16     v8h;
typedef __attribute__((ext_vector_type(2)))  _Float16     h2;
typedef __attribute__((ext_vector_type(8)))  float        v8f;
typedef __attribute__((ext_vector_type(4)))  unsigned int v4u;
typedef __attribute__((ext_vector_type(8)))  int          v8i;
typedef __attribute__((ext_vector_type(4)))  int          v4i;

#define S_FR   8
#define BATCH  2
#define CKD    64
#define CVD    64
#define HWD    4096
#define SHW    32768
#define NCHUNK 512          // SHW / 64 key-chunks
#define ST     72           // padded LDS row stride (halfs): conflict-free
#define LOG2E  1.4426950408889634f

__device__ __forceinline__ v16h cat8(v8h lo, v8h hi) {
  return __builtin_shufflevector(lo, hi, 0,1,2,3,4,5,6,7,8,9,10,11,12,13,14,15);
}
__device__ __forceinline__ v8f wmma16(v16h a, v16h b, v8f c) {
  return __builtin_amdgcn_wmma_f32_16x16x32_f16(false, a, false, b, (short)0, c,
                                                false, false);
}
__device__ __forceinline__ int h2i(h2 v) { return __builtin_bit_cast(int, v); }
__device__ __forceinline__ h2  i2h(int v) { return __builtin_bit_cast(h2, v); }

// ---------------------------------------------------------------------------
// TDM: load one 64-row x 128B 2-D f16 tile into LDS, padding +16B per row
// (row stride in LDS becomes 144B = the bank-conflict-free ST stride).
// D# fields per CDNA5 ISA.  This toolchain exposes
// the 6-arg builtin: (v4u g0, v8i g1, v4i, v4i, v8i, i32 cpol).
// ---------------------------------------------------------------------------
__device__ __forceinline__ void tdm_tile_load(unsigned lds_addr,
                                              const _Float16* g,
                                              unsigned stride_elems) {
  const unsigned long long ga = (unsigned long long)(uintptr_t)g;
  v4u g0;
  g0[0] = 1u;                                        // count=1 (valid D#)
  g0[1] = lds_addr;                                  // lds_addr [63:32]
  g0[2] = (unsigned)ga;                              // global_addr lo
  g0[3] = (unsigned)((ga >> 32) & 0x01FFFFFFull)     // global_addr [56:32]
          | (2u << 30);                              // type=2 ("image")
  v8i g1;
  g1[0] = (1 << 16)      // data_size = 1 -> 2 bytes
        | (1 << 20)      // pad_enable
        | (4 << 22)      // pad_interval: 2^(4+1)=32 DWORDs (128B) per pad
        | (3 << 25);     // pad_amount: 3+1 = 4 DWORDs (16B) of pad
  g1[1] = 64 << 16;      // tensor_dim0 = 64 elems (bits 79:48, low half)
  g1[2] = 64 << 16;      // tensor_dim0 hi=0 | tensor_dim1 = 64 (bits 111:80)
  g1[3] = 64 << 16;      // tensor_dim1 hi=0 | tile_dim0 = 64 (bits 127:112)
  g1[4] = 64;            // tile_dim1 = 64 | tile_dim2 = 0
  g1[5] = (int)stride_elems;  // tensor_dim0_stride[31:0]
  g1[6] = 0;             // stride hi | tensor_dim1_stride lo
  g1[7] = 0;
  v4i z4 = {0, 0, 0, 0};
  v8i z8 = {0, 0, 0, 0, 0, 0, 0, 0};
  __builtin_amdgcn_tensor_load_to_lds(g0, g1, z4, z4, z8, 0);
}

// ---------------------------------------------------------------------------
// Pre-pass A: maskvec[b][v] (post-softmax mask is row-constant -> one vector)
// ---------------------------------------------------------------------------
__global__ __launch_bounds__(256) void maskvec_kernel(
    const float* __restrict__ MV, const float* __restrict__ disp,
    const int* __restrict__ seq, float* __restrict__ maskvec) {
  const int b = blockIdx.x >> 6;
  const int v = blockIdx.x & 63;
  const int tid = threadIdx.x;
  __shared__ float red[256];
  float partial = 0.f;
  const float* drow = disp + (size_t)b * HWD;
  for (int s = 0; s < S_FR; ++s) {
    const float dy = (float)seq[(b * S_FR + s) * 2 + 1] - 5.0f;
    const float dx = (float)seq[(b * S_FR + s) * 2 + 0] - 5.0f;
    const float dist = sqrtf(dy * dy + dx * dx);
    const float* vrow = MV + ((size_t)(s * BATCH + b) * CVD + v) * HWD;
    for (int hw = tid; hw < HWD; hw += 256)
      if (fabsf(dist * drow[hw]) > 0.1f) partial += vrow[hw];
  }
  red[tid] = partial;
  __syncthreads();
  for (int off = 128; off > 0; off >>= 1) {
    if (tid < off) red[tid] += red[tid + off];
    __syncthreads();
  }
  if (tid == 0) maskvec[b * CVD + v] = red[0] * (0.2f / 32768.0f);
}

// ---------------------------------------------------------------------------
// Pre-pass B: K f32 [S,B,Ck,H,W] -> f16 Khalf [b][m][c] (transpose via LDS)
// ---------------------------------------------------------------------------
__global__ __launch_bounds__(256) void convK_kernel(
    const float* __restrict__ MK, _Float16* __restrict__ Khalf) {
  const int b   = blockIdx.x >> 9;        // grid = 2*8*64
  const int rem = blockIdx.x & 511;
  const int s   = rem >> 6;
  const int hw0 = (rem & 63) << 6;
  __shared__ __align__(16) _Float16 tile[64 * ST];  // [m][c]
  const int tid = threadIdx.x;
  const float* src = MK + (size_t)(s * BATCH + b) * CKD * HWD + hw0;
#pragma unroll
  for (int it = 0; it < 4; ++it) {
    const int c  = it * 16 + (tid >> 4);
    const int m4 = (tid & 15) * 4;
    const float4 kv = *(const float4*)(src + (size_t)c * HWD + m4);
    tile[(m4 + 0) * ST + c] = (_Float16)kv.x;
    tile[(m4 + 1) * ST + c] = (_Float16)kv.y;
    tile[(m4 + 2) * ST + c] = (_Float16)kv.z;
    tile[(m4 + 3) * ST + c] = (_Float16)kv.w;
  }
  __syncthreads();
  _Float16* dst = Khalf + ((size_t)b * SHW + (size_t)s * HWD + hw0) * CKD;
#pragma unroll
  for (int i = 0; i < 2; ++i) {
    const int chunk = i * 256 + tid;      // 512 chunks of 8 halfs
    const int m = chunk >> 3;
    const int o = (chunk & 7) * 8;
    *(v8h*)(dst + (size_t)m * CKD + o) = *(const v8h*)(tile + m * ST + o);
  }
}

// ---------------------------------------------------------------------------
// Pre-pass C: V f32 [S,B,Cv,H,W] -> f16 Vhalf [b][v][m] (pure convert/copy)
// ---------------------------------------------------------------------------
__global__ __launch_bounds__(256) void convV_kernel(
    const float* __restrict__ MV, _Float16* __restrict__ Vhalf) {
  const size_t qd  = (size_t)blockIdx.x * 256 + threadIdx.x;  // quad index
  const int hwq = (int)(qd & 1023);
  const int s   = (int)(qd >> 10) & 7;
  const int v   = (int)(qd >> 13) & 63;
  const int b   = (int)(qd >> 19);
  const float4 x =
      *(const float4*)(MV + ((size_t)(s * BATCH + b) * CVD + v) * HWD + hwq * 4);
  h2 d0; d0.x = (_Float16)x.x; d0.y = (_Float16)x.y;
  h2 d1; d1.x = (_Float16)x.z; d1.y = (_Float16)x.w;
  h2* o = (h2*)(Vhalf + ((size_t)b * CVD + v) * SHW + (size_t)s * HWD + hwq * 4);
  o[0] = d0; o[1] = d1;
}

// ---------------------------------------------------------------------------
// Flash attention, transposed (S^T = K*Q^T, out^T = V^T*P^T), wave32 WMMA.
// USE_TDM: double-buffered TENSOR_LOAD_TO_LDS staging from pre-converted f16.
// else:    in-kernel f32->f16 convert+transpose staging (fallback).
// ---------------------------------------------------------------------------
template <bool USE_TDM>
__global__ __launch_bounds__(128) void fattn_kernel(
    const float* __restrict__ MK, const float* __restrict__ MV,
    const float* __restrict__ QQ, const _Float16* __restrict__ Khalf,
    const _Float16* __restrict__ Vhalf, const float* __restrict__ maskvec,
    float* __restrict__ out) {
  __shared__ __align__(16) _Float16 ldsK[2][64 * ST];  // [key m][feat c]
  __shared__ __align__(16) _Float16 ldsV[2][64 * ST];  // [feat v][key m]

  const int tid  = threadIdx.x;
  const int b    = blockIdx.x >> 6;
  const int qb   = (blockIdx.x & 63) * 64;
  const int wave = tid >> 5;
  const int lane = tid & 31;
  const int l16  = lane & 15;
  const int hi   = lane >> 4;
  const int qcol = qb + wave * 16 + l16;

  // Q^T as persistent B-operands: lane -> query col, half e -> feature
  v16h qB0, qB1;
  {
    const float* qp = QQ + (size_t)b * CKD * HWD + qcol;
#pragma unroll
    for (int e = 0; e < 16; ++e) {
      qB0[e] = (_Float16)qp[(size_t)(hi * 8 + e + (e >= 8 ? 8 : 0)) * HWD];
      qB1[e] = (_Float16)qp[(size_t)(32 + hi * 8 + e + (e >= 8 ? 8 : 0)) * HWD];
    }
  }

  v8f acc[4] = {};
  float lsum = 0.f;
  float mrow = -INFINITY;

  const int rr = tid >> 4;
  const int n4 = (tid & 15) * 4;

  const _Float16* Kb = Khalf + (size_t)b * SHW * CKD;
  const _Float16* Vb = Vhalf + (size_t)b * CVD * SHW;
  unsigned aK[2], aV[2];
  if constexpr (USE_TDM) {
    aK[0] = (unsigned)(uintptr_t)&ldsK[0][0];
    aK[1] = (unsigned)(uintptr_t)&ldsK[1][0];
    aV[0] = (unsigned)(uintptr_t)&ldsV[0][0];
    aV[1] = (unsigned)(uintptr_t)&ldsV[1][0];
    if (wave == 0) {
      tdm_tile_load(aK[0], Kb, 64);          // chunk 0: K 64x64, row stride 64
      tdm_tile_load(aV[0], Vb, SHW);         // chunk 0: V^T 64x64, stride 32768
      __builtin_amdgcn_s_wait_tensorcnt(0);
    }
    __syncthreads();
  }

#pragma unroll 1
  for (int kc = 0; kc < NCHUNK; ++kc) {
    const int cur = USE_TDM ? (kc & 1) : 0;

    if constexpr (USE_TDM) {
      if (wave == 0 && kc + 1 < NCHUNK) {    // overlap DMA of next chunk
        const size_t m1 = (size_t)(kc + 1) * 64;
        tdm_tile_load(aK[1 - cur], Kb + m1 * CKD, 64);
        tdm_tile_load(aV[1 - cur], Vb + m1, SHW);
      }
    } else {
      const int s   = kc >> 6;
      const int hw0 = (kc & 63) << 6;
      const float* kg = MK + (size_t)(s * BATCH + b) * CKD * HWD + hw0;
      const float* vg = MV + (size_t)(s * BATCH + b) * CVD * HWD + hw0;
      __syncthreads();
#pragma unroll
      for (int it = 0; it < 8; ++it) {
        const int c = it * 8 + rr;
        const float4 kv = *(const float4*)(kg + (size_t)c * HWD + n4);
        ldsK[0][(n4 + 0) * ST + c] = (_Float16)kv.x;
        ldsK[0][(n4 + 1) * ST + c] = (_Float16)kv.y;
        ldsK[0][(n4 + 2) * ST + c] = (_Float16)kv.z;
        ldsK[0][(n4 + 3) * ST + c] = (_Float16)kv.w;
        const float4 vv = *(const float4*)(vg + (size_t)c * HWD + n4);
        h2* dst = (h2*)(&ldsV[0][c * ST + n4]);
        h2 d0; d0.x = (_Float16)vv.x; d0.y = (_Float16)vv.y;
        h2 d1; d1.x = (_Float16)vv.z; d1.y = (_Float16)vv.w;
        dst[0] = d0; dst[1] = d1;
      }
      if (kc + 1 < NCHUNK) {
        const int s2 = (kc + 1) >> 6, hw2 = ((kc + 1) & 63) << 6;
        const size_t roff = (size_t)(tid >> 1) * HWD + (tid & 1) * 32;
        __builtin_prefetch(MK + (size_t)(s2 * BATCH + b) * CKD * HWD + hw2 + roff, 0, 3);
        __builtin_prefetch(MV + (size_t)(s2 * BATCH + b) * CVD * HWD + hw2 + roff, 0, 3);
      }
      __syncthreads();
    }

    // S^T = K(16x64) * Q^T(64x16), 4 key tiles; lane holds one query column.
    v8f stt[4];
#pragma unroll
    for (int t = 0; t < 4; ++t) {
      const _Float16* kr = &ldsK[cur][(t * 16 + l16) * ST + hi * 8];
      const v16h a0 = cat8(*(const v8h*)(kr + 0),  *(const v8h*)(kr + 16));
      const v16h a1 = cat8(*(const v8h*)(kr + 32), *(const v8h*)(kr + 48));
      v8f z = {};
      v8f c = wmma16(a0, qB0, z);
      c = wmma16(a1, qB1, c);
      stt[t] = c;
    }

    // Online softmax: in-lane over 32 keys + one partner-half shuffle.
    float cmax = -INFINITY;
#pragma unroll
    for (int t = 0; t < 4; ++t)
#pragma unroll
      for (int j = 0; j < 8; ++j) cmax = fmaxf(cmax, stt[t][j]);
    cmax = fmaxf(cmax, __shfl_xor(cmax, 16, 32));
    const float mnew  = fmaxf(mrow, cmax);
    const float scale = __builtin_amdgcn_exp2f((mrow - mnew) * LOG2E);
    mrow = mnew;
    lsum *= scale;
#pragma unroll
    for (int t = 0; t < 4; ++t)
#pragma unroll
      for (int j = 0; j < 8; ++j) acc[t][j] *= scale;

    h2 own[4][4];
#pragma unroll
    for (int t = 0; t < 4; ++t) {
#pragma unroll
      for (int j = 0; j < 4; ++j) {
        const float p0 = __builtin_amdgcn_exp2f((stt[t][2 * j]     - mnew) * LOG2E);
        const float p1 = __builtin_amdgcn_exp2f((stt[t][2 * j + 1] - mnew) * LOG2E);
        lsum += p0 + p1;
        h2 h; h.x = (_Float16)p0; h.y = (_Float16)p1;
        own[t][j] = h;
      }
    }

    // Relayout P^T (C-layout) -> B-operand layout via partner-lane exchange.
    v16h pB0, pB1;
#pragma unroll
    for (int j = 0; j < 4; ++j) {
      const int p0 = __shfl_xor(h2i(own[0][j]), 16, 32);
      const int p1 = __shfl_xor(h2i(own[1][j]), 16, 32);
      const int p2 = __shfl_xor(h2i(own[2][j]), 16, 32);
      const int p3 = __shfl_xor(h2i(own[3][j]), 16, 32);
      const h2 lo0 = i2h(hi ? p1 : h2i(own[0][j]));
      const h2 hi0 = i2h(hi ? h2i(own[1][j]) : p0);
      const h2 lo1 = i2h(hi ? p3 : h2i(own[2][j]));
      const h2 hi1 = i2h(hi ? h2i(own[3][j]) : p2);
      pB0[2 * j]     = lo0.x; pB0[2 * j + 1]     = lo0.y;
      pB0[8 + 2 * j] = hi0.x; pB0[8 + 2 * j + 1] = hi0.y;
      pB1[2 * j]     = lo1.x; pB1[2 * j + 1]     = lo1.y;
      pB1[8 + 2 * j] = hi1.x; pB1[8 + 2 * j + 1] = hi1.y;
    }

    // out^T += V^T(16x64) * P^T(64x16) per v-tile.
#pragma unroll
    for (int vt = 0; vt < 4; ++vt) {
      const _Float16* vr = &ldsV[cur][(vt * 16 + l16) * ST + hi * 16];
      const v16h a0 = cat8(*(const v8h*)(vr + 0),  *(const v8h*)(vr + 8));
      const v16h a1 = cat8(*(const v8h*)(vr + 32), *(const v8h*)(vr + 40));
      acc[vt] = wmma16(a0, pB0, acc[vt]);
      acc[vt] = wmma16(a1, pB1, acc[vt]);
    }

    if constexpr (USE_TDM) {
      if (wave == 0) __builtin_amdgcn_s_wait_tensorcnt(0);
      __syncthreads();
    }
  }

  // Epilogue: merge partner denominator, normalize, add mask*V, store out^T.
  lsum += __shfl_xor(lsum, 16, 32);
  const float inv = 1.0f / lsum;
  float* ob = out + (size_t)b * CVD * HWD + qcol;
  const float* mvp = maskvec + b * CVD;
#pragma unroll
  for (int vt = 0; vt < 4; ++vt) {
#pragma unroll
    for (int j = 0; j < 8; ++j) {
      const int v = vt * 16 + hi * 8 + j;
      ob[(size_t)v * HWD] = acc[vt][j] * inv + mvp[v];
    }
  }
}

extern "C" void kernel_launch(void* const* d_in, const int* in_sizes, int n_in,
                              void* d_out, int out_size, void* d_ws, size_t ws_size,
                              hipStream_t stream) {
  const float* MK   = (const float*)d_in[0];  // memory_keys   [S,B,Ck,H,W]
  const float* MV   = (const float*)d_in[1];  // memory_values [S,B,Cv,H,W]
  const float* QQ   = (const float*)d_in[2];  // query_query   [B,Ck,H,W]
  const float* disp = (const float*)d_in[3];  // disparity     [B,1,H,W]
  const int*   seq  = (const int*)d_in[4];    // sequence_index [B,S,2]
  float* outp = (float*)d_out;

  float* maskvec = (float*)d_ws;              // 128 floats @ offset 0
  const size_t KOFF = 4096;
  const size_t VOFF = KOFF + (size_t)BATCH * SHW * CKD * sizeof(_Float16);
  const size_t NEED = VOFF + (size_t)BATCH * CVD * SHW * sizeof(_Float16);

  maskvec_kernel<<<BATCH * CVD, 256, 0, stream>>>(MV, disp, seq, maskvec);

  if (ws_size >= NEED) {
    _Float16* Khalf = (_Float16*)((char*)d_ws + KOFF);
    _Float16* Vhalf = (_Float16*)((char*)d_ws + VOFF);
    convK_kernel<<<BATCH * S_FR * (HWD / 64), 256, 0, stream>>>(MK, Khalf);
    convV_kernel<<<(BATCH * CVD * SHW) / (4 * 256), 256, 0, stream>>>(MV, Vhalf);
    fattn_kernel<true><<<BATCH * (HWD / 64), 128, 0, stream>>>(
        MK, MV, QQ, Khalf, Vhalf, maskvec, outp);
  } else {
    fattn_kernel<false><<<BATCH * (HWD / 64), 128, 0, stream>>>(
        MK, MV, QQ, nullptr, nullptr, maskvec, outp);
  }
}
